// AttentionBlock_48885317763171
// MI455X (gfx1250) — matrix operand order repaired and checked by reading, not compiled, not run on hardware
//
#include <hip/hip_runtime.h>
#include <hip/hip_bf16.h>

typedef __attribute__((ext_vector_type(16))) __bf16 v16bf;
typedef __attribute__((ext_vector_type(8)))  float  v8f;

// ---------- helpers ----------
__device__ __forceinline__ unsigned short f2bf(float f) {
  unsigned int u = __builtin_bit_cast(unsigned int, f);
  return (unsigned short)((u + 0x7fffu + ((u >> 16) & 1u)) >> 16);  // RNE
}
__device__ __forceinline__ __bf16 bfbits(unsigned short s) {
  return __builtin_bit_cast(__bf16, s);
}

// A-operand 16x32 slice from row-major [M, ld] bf16.
// ISA layout: lane(m=lane&15, hs=lane>>4); halfs j<8 -> k=j+8*hs, j>=8 -> k=16+(j-8)+8*hs
__device__ __forceinline__ v16bf load_a32(const unsigned short* base, int ld) {
  int lane = threadIdx.x & 31;
  int m = lane & 15, hs = lane >> 4;
  const unsigned short* p = base + m * ld + 8 * hs;
  v16bf f;
#pragma unroll
  for (int j = 0; j < 8; ++j) {
    f[j]     = bfbits(p[j]);
    f[j + 8] = bfbits(p[16 + j]);
  }
  return f;
}

// B-operand 32x16 from row-major [N, ld] bf16 (B^T in memory).
// ISA layout: lane(n=lane&15, hs); halfs j<8 -> k=j+8*hs, j>=8 -> k=16+(j-8)+8*hs
__device__ __forceinline__ v16bf load_bt32(const unsigned short* base, int ld) {
  int lane = threadIdx.x & 31;
  int n = lane & 15, hs = lane >> 4;
  const unsigned short* p = base + n * ld + 8 * hs;
  v16bf f;
#pragma unroll
  for (int j = 0; j < 8; ++j) {
    f[j]     = bfbits(p[j]);
    f[j + 8] = bfbits(p[16 + j]);
  }
  return f;
}

__device__ __forceinline__ v8f wmma_bf16(v16bf a, v16bf b, v8f c) {
  return __builtin_amdgcn_wmma_f32_16x16x32_bf16(false, a, false, b, (short)0, c,
                                                 false, false);
}

// ---------- prep: x[B,C,S] f32 -> xs[B*S, C] bf16 (LDS tiled transpose) ----------
__global__ __launch_bounds__(256) void x_to_xs(const float* __restrict__ x,
                                               unsigned short* __restrict__ xs) {
  __shared__ float tile[32][33];
  int b = blockIdx.z;
  int s0 = blockIdx.x * 32, c0 = blockIdx.y * 32;
  int tx = threadIdx.x, ty = threadIdx.y;
#pragma unroll
  for (int i = 0; i < 4; ++i) {
    int cc = ty + i * 8;
    tile[cc][tx] = x[((size_t)(b * 256 + c0 + cc)) * 1024 + s0 + tx];
  }
  __syncthreads();
#pragma unroll
  for (int i = 0; i < 4; ++i) {
    int ss = ty + i * 8;
    xs[((size_t)(b * 1024 + s0 + ss)) * 256 + c0 + tx] = f2bf(tile[tx][ss]);
  }
}

// ---------- prep: w[K,N] f32 -> wT[N,K] bf16 ----------
__global__ void wt_bf16(const float* __restrict__ w, unsigned short* __restrict__ wT,
                        int K, int N) {
  int tid = blockIdx.x * blockDim.x + threadIdx.x;
  if (tid >= K * N) return;
  int n = tid / K, kk = tid - n * K;
  wT[tid] = f2bf(w[(size_t)kk * N + n]);
}

// ---------- QKV GEMM: qkv = xs @ Wp + b, scatter to q[S,d], k[S,d], vT[d,S] ----------
// Wave tile 32x64 (2 A-frags x 4 B-frags = 8 WMMA / k-step), software pipelined.
__global__ __launch_bounds__(256) void qkv_gemm(const unsigned short* __restrict__ xs,
                                                const unsigned short* __restrict__ wpT,
                                                const float* __restrict__ bproj,
                                                unsigned short* __restrict__ q,
                                                unsigned short* __restrict__ kk,
                                                unsigned short* __restrict__ vT) {
  const int lane = threadIdx.x & 31;
  const int wv = threadIdx.x >> 5;
  const int nl = lane & 15, hs = lane >> 4;
  const int m0 = blockIdx.x * 256 + wv * 32;
  const int n0 = blockIdx.y * 64;

  v8f acc[8];  // [mi][c]
#pragma unroll
  for (int c = 0; c < 4; ++c) {
    float bv = bproj[n0 + c * 16 + nl];
#pragma unroll
    for (int r = 0; r < 8; ++r) { acc[c][r] = bv; acc[4 + c][r] = bv; }
  }

  const unsigned short* arow0 = xs + (size_t)m0 * 256;
  const unsigned short* arow1 = xs + (size_t)(m0 + 16) * 256;
  v16bf a0 = load_a32(arow0, 256);
  v16bf a1 = load_a32(arow1, 256);
  v16bf bfr[4];
#pragma unroll
  for (int c = 0; c < 4; ++c) bfr[c] = load_bt32(wpT + (size_t)(n0 + c * 16) * 256, 256);

  for (int k0 = 0; k0 < 256; k0 += 32) {
    v16bf a0c = a0, a1c = a1;
    v16bf b0c = bfr[0], b1c = bfr[1], b2c = bfr[2], b3c = bfr[3];
    if (k0 + 32 < 256) {  // issue next k-step's loads before consuming current frags
      a0 = load_a32(arow0 + k0 + 32, 256);
      a1 = load_a32(arow1 + k0 + 32, 256);
#pragma unroll
      for (int c = 0; c < 4; ++c)
        bfr[c] = load_bt32(wpT + (size_t)(n0 + c * 16) * 256 + k0 + 32, 256);
    }
    acc[0] = wmma_bf16(a0c, b0c, acc[0]);
    acc[1] = wmma_bf16(a0c, b1c, acc[1]);
    acc[2] = wmma_bf16(a0c, b2c, acc[2]);
    acc[3] = wmma_bf16(a0c, b3c, acc[3]);
    acc[4] = wmma_bf16(a1c, b0c, acc[4]);
    acc[5] = wmma_bf16(a1c, b1c, acc[5]);
    acc[6] = wmma_bf16(a1c, b2c, acc[6]);
    acc[7] = wmma_bf16(a1c, b3c, acc[7]);
  }

#pragma unroll
  for (int mi = 0; mi < 2; ++mi) {
#pragma unroll
    for (int c = 0; c < 4; ++c) {
      int ng = n0 + c * 16 + nl;
      int h = ng / 192;
      int t = ng - h * 192;
      int seg = t >> 6;  // 0=q 1=k 2=v
      int d = t & 63;
#pragma unroll
      for (int r = 0; r < 8; ++r) {
        int mg = m0 + mi * 16 + r + 8 * hs;
        int b = mg >> 10, s = mg & 1023;
        unsigned short v = f2bf(acc[mi * 4 + c][r]);
        size_t bh = (size_t)(b * 4 + h);
        if (seg == 0)      q [(bh * 1024 + s) * 64 + d] = v;
        else if (seg == 1) kk[(bh * 1024 + s) * 64 + d] = v;
        else               vT[(bh * 64 + d) * 1024 + s] = v;  // V stored transposed
      }
    }
  }
}

// ---------- flash attention: one 16-row strip per wave, online softmax ----------
// Pipelined: K-frags preloaded one j-tile ahead; V loads issued before the softmax
// VALU block so v_exp/shuffle latency covers them.
__global__ __launch_bounds__(256) void attn_fa(const unsigned short* __restrict__ q,
                                               const unsigned short* __restrict__ kk,
                                               const unsigned short* __restrict__ vT,
                                               unsigned short* __restrict__ res) {
  __shared__ unsigned short pbuf[8][16 * 32];  // per-wave P staging (1KB each)
  const int lane = threadIdx.x & 31;
  const int wv = threadIdx.x >> 5;
  const int nl = lane & 15, hs = lane >> 4;
  const int gw = blockIdx.x * 8 + wv;  // 4096 wave-tiles total
  const int itile = gw & 63;
  const int bh = gw >> 6;
  const int b = bh >> 2, h = bh & 3;
  const int i0 = itile * 16;

  const unsigned short* qb = q  + (size_t)bh * 1024 * 64 + (size_t)i0 * 64;
  const unsigned short* kb = kk + (size_t)bh * 1024 * 64;
  const unsigned short* vb = vT + (size_t)bh * 64 * 1024;

  v16bf aq0 = load_a32(qb + 0, 64);
  v16bf aq1 = load_a32(qb + 32, 64);

  float mrow[8], lrow[8];
  v8f o0 = {}, o1 = {}, o2 = {}, o3 = {};
#pragma unroll
  for (int r = 0; r < 8; ++r) { mrow[r] = -3.0e38f; lrow[r] = 0.f; }

  unsigned short* pb = &pbuf[wv][0];

  // preload K fragments for j0 = 0
  v16bf kf0 = load_bt32(kb + 0 * 64 + 0, 64);
  v16bf kf1 = load_bt32(kb + 0 * 64 + 32, 64);
  v16bf kf2 = load_bt32(kb + 16 * 64 + 0, 64);
  v16bf kf3 = load_bt32(kb + 16 * 64 + 32, 64);

  for (int j0 = 0; j0 < 1024; j0 += 32) {
    // scores for two 16x16 j-tiles (K = d = 64, two chained WMMAs each)
    v8f s0 = {}, s1 = {};
    s0 = wmma_bf16(aq0, kf0, s0);
    s0 = wmma_bf16(aq1, kf1, s0);
    s1 = wmma_bf16(aq0, kf2, s1);
    s1 = wmma_bf16(aq1, kf3, s1);

    // issue V loads now: softmax VALU below hides their latency
    v16bf vf0 = load_bt32(vb + (size_t)0 * 16 * 1024 + j0, 1024);
    v16bf vf1 = load_bt32(vb + (size_t)1 * 16 * 1024 + j0, 1024);
    v16bf vf2 = load_bt32(vb + (size_t)2 * 16 * 1024 + j0, 1024);
    v16bf vf3 = load_bt32(vb + (size_t)3 * 16 * 1024 + j0, 1024);

    // prefetch next j-tile's K fragments
    if (j0 + 32 < 1024) {
      kf0 = load_bt32(kb + (size_t)(j0 + 32) * 64 + 0, 64);
      kf1 = load_bt32(kb + (size_t)(j0 + 32) * 64 + 32, 64);
      kf2 = load_bt32(kb + (size_t)(j0 + 48) * 64 + 0, 64);
      kf3 = load_bt32(kb + (size_t)(j0 + 48) * 64 + 32, 64);
    }

#pragma unroll
    for (int r = 0; r < 8; ++r) {
      float a = s0[r] * 0.125f;  // 1/sqrt(64)
      float c = s1[r] * 0.125f;
      float mx = fmaxf(a, c);
#pragma unroll
      for (int msk = 1; msk < 16; msk <<= 1) mx = fmaxf(mx, __shfl_xor(mx, msk, 32));
      float mn = fmaxf(mrow[r], mx);
      float alpha = __expf(mrow[r] - mn);
      float e0 = __expf(a - mn);
      float e1 = __expf(c - mn);
      float rs = e0 + e1;
#pragma unroll
      for (int msk = 1; msk < 16; msk <<= 1) rs += __shfl_xor(rs, msk, 32);
      lrow[r] = lrow[r] * alpha + rs;
      mrow[r] = mn;
      o0[r] *= alpha; o1[r] *= alpha; o2[r] *= alpha; o3[r] *= alpha;
      int row = r + 8 * hs;  // C-layout row owned by this half-wave
      pb[row * 32 + nl]      = f2bf(e0);
      pb[row * 32 + 16 + nl] = f2bf(e1);
    }
    // same-wave DS ops are in-order; wait for all 32 lanes' P writes to land
    asm volatile("s_wait_dscnt 0" ::: "memory");
    v16bf pf = load_a32(pb, 32);  // re-read P in A-operand layout
    o0 = wmma_bf16(pf, vf0, o0);
    o1 = wmma_bf16(pf, vf1, o1);
    o2 = wmma_bf16(pf, vf2, o2);
    o3 = wmma_bf16(pf, vf3, o3);
  }

  unsigned short* rbase = res + ((size_t)(b * 1024 + i0)) * 256 + h * 64;
#pragma unroll
  for (int r = 0; r < 8; ++r) {
    int row = r + 8 * hs;
    float inv = 1.0f / lrow[r];
    rbase[row * 256 +  0 + nl] = f2bf(o0[r] * inv);
    rbase[row * 256 + 16 + nl] = f2bf(o1[r] * inv);
    rbase[row * 256 + 32 + nl] = f2bf(o2[r] * inv);
    rbase[row * 256 + 48 + nl] = f2bf(o3[r] * inv);
  }
}

// ---------- out proj: out[b,c,s] = res @ Wo + b_out + x (residual) ----------
__global__ __launch_bounds__(256) void out_gemm(const unsigned short* __restrict__ res,
                                                const unsigned short* __restrict__ woT,
                                                const float* __restrict__ bout,
                                                const float* __restrict__ x,
                                                float* __restrict__ out) {
  const int lane = threadIdx.x & 31;
  const int wv = threadIdx.x >> 5;
  const int nl = lane & 15, hs = lane >> 4;
  const int m0 = blockIdx.x * 256 + wv * 32;
  const int n0 = blockIdx.y * 64;

  v8f acc[8];
#pragma unroll
  for (int c = 0; c < 4; ++c) {
    float bv = bout[n0 + c * 16 + nl];
#pragma unroll
    for (int r = 0; r < 8; ++r) { acc[c][r] = bv; acc[4 + c][r] = bv; }
  }

  const unsigned short* arow0 = res + (size_t)m0 * 256;
  const unsigned short* arow1 = res + (size_t)(m0 + 16) * 256;
  v16bf a0 = load_a32(arow0, 256);
  v16bf a1 = load_a32(arow1, 256);
  v16bf bfr[4];
#pragma unroll
  for (int c = 0; c < 4; ++c) bfr[c] = load_bt32(woT + (size_t)(n0 + c * 16) * 256, 256);

  for (int k0 = 0; k0 < 256; k0 += 32) {
    v16bf a0c = a0, a1c = a1;
    v16bf b0c = bfr[0], b1c = bfr[1], b2c = bfr[2], b3c = bfr[3];
    if (k0 + 32 < 256) {
      a0 = load_a32(arow0 + k0 + 32, 256);
      a1 = load_a32(arow1 + k0 + 32, 256);
#pragma unroll
      for (int c = 0; c < 4; ++c)
        bfr[c] = load_bt32(woT + (size_t)(n0 + c * 16) * 256 + k0 + 32, 256);
    }
    acc[0] = wmma_bf16(a0c, b0c, acc[0]);
    acc[1] = wmma_bf16(a0c, b1c, acc[1]);
    acc[2] = wmma_bf16(a0c, b2c, acc[2]);
    acc[3] = wmma_bf16(a0c, b3c, acc[3]);
    acc[4] = wmma_bf16(a1c, b0c, acc[4]);
    acc[5] = wmma_bf16(a1c, b1c, acc[5]);
    acc[6] = wmma_bf16(a1c, b2c, acc[6]);
    acc[7] = wmma_bf16(a1c, b3c, acc[7]);
  }

#pragma unroll
  for (int mi = 0; mi < 2; ++mi) {
#pragma unroll
    for (int c = 0; c < 4; ++c) {
      int n = n0 + c * 16 + nl;
#pragma unroll
      for (int r = 0; r < 8; ++r) {
        int mg = m0 + mi * 16 + r + 8 * hs;
        int b = mg >> 10, s = mg & 1023;
        size_t idx = ((size_t)(b * 256 + n)) * 1024 + s;  // [B,C,S] out == x layout
        out[idx] = acc[mi * 4 + c][r] + x[idx];
      }
    }
  }
}

// ---------- launch ----------
extern "C" void kernel_launch(void* const* d_in, const int* in_sizes, int n_in,
                              void* d_out, int out_size, void* d_ws, size_t ws_size,
                              hipStream_t stream) {
  const float* x      = (const float*)d_in[0];
  const float* w_proj = (const float*)d_in[1];
  const float* b_proj = (const float*)d_in[2];
  const float* w_out  = (const float*)d_in[3];
  const float* b_out  = (const float*)d_in[4];
  float* out = (float*)d_out;

  char* ws = (char*)d_ws;
  unsigned short* xs  = (unsigned short*)(ws);
  unsigned short* q   = (unsigned short*)(ws + (size_t)8  * 1024 * 1024);
  unsigned short* k   = (unsigned short*)(ws + (size_t)16 * 1024 * 1024);
  unsigned short* vT  = (unsigned short*)(ws + (size_t)24 * 1024 * 1024);
  unsigned short* res = (unsigned short*)(ws + (size_t)32 * 1024 * 1024);
  unsigned short* wpT = (unsigned short*)(ws + (size_t)40 * 1024 * 1024);
  unsigned short* woT = (unsigned short*)(ws + (size_t)40 * 1024 * 1024 + 768 * 256 * 2);

  x_to_xs<<<dim3(32, 8, 16), dim3(32, 8), 0, stream>>>(x, xs);
  wt_bf16<<<(768 * 256 + 255) / 256, 256, 0, stream>>>(w_proj, wpT, 256, 768);
  wt_bf16<<<(256 * 256 + 255) / 256, 256, 0, stream>>>(w_out, woT, 256, 256);
  qkv_gemm<<<dim3(64, 12), 256, 0, stream>>>(xs, wpT, b_proj, q, k, vT);
  attn_fa<<<512, 256, 0, stream>>>(q, k, vT, res);
  out_gemm<<<dim3(64, 4), 256, 0, stream>>>(res, woT, b_out, x, out);
}
